// SimpleGNN_75840532513258
// MI455X (gfx1250) — matrix operand order repaired and checked by reading, not compiled, not run on hardware
//
#include <hip/hip_runtime.h>
#include <hip/hip_bf16.h>

typedef _Float16 half_t;
typedef __attribute__((ext_vector_type(16))) _Float16 v16h;
typedef __attribute__((ext_vector_type(8)))  _Float16 v8h;
typedef __attribute__((ext_vector_type(8)))  float    v8f;

#define H_DIM 128

// ---------------------------------------------------------------------------
// Convert fp32 weight [K,N] -> TRANSPOSED padded fp16 BT[128][KPad] (zero fill)
// ---------------------------------------------------------------------------
__global__ void k_convert_pad_T(const float* __restrict__ W, half_t* __restrict__ BT,
                                int K, int N, int KPad) {
    int idx = blockIdx.x * blockDim.x + threadIdx.x;
    int tot = KPad * H_DIM;
    if (idx >= tot) return;
    int n = idx / KPad, k = idx % KPad;
    float v = (k < K && n < N) ? W[k * N + n] : 0.0f;
    BT[idx] = (half_t)v;
}

// ---------------------------------------------------------------------------
// Per-edge input gather: A[e, :] = concat(v[dst]+v[src], edge_attr[e], 0-pad)
// ---------------------------------------------------------------------------
__global__ void k_gather(const float* __restrict__ v, int Dv,
                         const float* __restrict__ ea,
                         const int* __restrict__ ei, int E_,
                         half_t* __restrict__ A, int DinPad) {
    int e = blockIdx.x;
    int c = threadIdx.x;
    int srcN = ei[e];
    int dstN = ei[E_ + e];
    float val;
    if (c < Dv)            val = v[(size_t)dstN * Dv + c] + v[(size_t)srcN * Dv + c];
    else if (c < Dv + 16)  val = ea[(size_t)e * 16 + (c - Dv)];
    else                   val = 0.0f;
    A[(size_t)e * DinPad + c] = (half_t)val;
}

// ---------------------------------------------------------------------------
__global__ void k_deg(const int* __restrict__ ei, int E_, float* __restrict__ deg) {
    int e = blockIdx.x * blockDim.x + threadIdx.x;
    if (e < E_) atomicAdd(&deg[ei[E_ + e]], 1.0f);
}

// ---------------------------------------------------------------------------
// Tiled WMMA GEMM:  C[M,128] = A[M,Kd] (f16) x B (f16, pre-transposed BT[128][Kd])
//   REQUIRES M % 128 == 0 (E = 320000 = 2500 * 128).
//   256 threads = 8 waves; block tile 128x128; wave tile 32x64 (2x4 WMMA tiles).
//   Double-buffered LDS: global loads for tile k+1 issue before the WMMAs of
//   tile k; single barrier per K-step (ping-pong covers both hazards).
// ---------------------------------------------------------------------------
__global__ __launch_bounds__(256)
void k_gemm(const half_t* __restrict__ A, int Kd,
            const half_t* __restrict__ BT, int Nreal,
            const float* __restrict__ bias,
            int relu, int scatter,
            half_t* __restrict__ Cout,
            float* __restrict__ acc, int accLd,
            const int* __restrict__ dstIdx) {
    __shared__ half_t As[2][128][40];    // 128 rows x 32 K, stride 40 (16B-aligned rows)
    __shared__ half_t BsT[2][128][40];   // 128 cols x 32 K

    const int tid  = threadIdx.x;
    const int lane = tid & 31;
    const int wid  = tid >> 5;
    const int wm   = (wid >> 1) * 32;   // wave row offset: 0,32,64,96
    const int wn   = (wid & 1) * 64;    // wave col offset: 0 or 64
    const size_t rowBase = (size_t)blockIdx.x * 128;

    // staging coordinates: 512 16B chunks per tile, 2 per thread
    int sr[2], sc[2];
#pragma unroll
    for (int it = 0; it < 2; ++it) {
        int chunk = tid + it * 256;
        sr[it] = chunk >> 2;            // row/col index 0..127
        sc[it] = (chunk & 3) * 8;       // K offset 0,8,16,24
    }

    v8f accf[2][4];
#pragma unroll
    for (int mt = 0; mt < 2; ++mt)
#pragma unroll
        for (int nt = 0; nt < 4; ++nt) { v8f z = {}; accf[mt][nt] = z; }

    const int r16  = lane & 15;
    const int koff = (lane >> 4) * 8;   // A frag: lanes 0-15 -> K 0..7,  16-31 -> K 8..15
    const int kb   = (lane >> 4) * 16;  // B frag uses koff (same K order as A)

    // ---- prologue: stage tile 0 ----
#pragma unroll
    for (int it = 0; it < 2; ++it) {
        *(v8h*)&As[0][sr[it]][sc[it]]  = *(const v8h*)&A [(rowBase + sr[it]) * (size_t)Kd + sc[it]];
        *(v8h*)&BsT[0][sr[it]][sc[it]] = *(const v8h*)&BT[(size_t)sr[it] * Kd + sc[it]];
    }
    __syncthreads();

    int buf = 0;
    for (int k0 = 0; k0 < Kd; k0 += 32, buf ^= 1) {
        const bool hasNext = (k0 + 32) < Kd;

        // ---- issue global loads for next tile (latency hidden by WMMAs) ----
        v8h ra[2], rb[2];
        if (hasNext) {
#pragma unroll
            for (int it = 0; it < 2; ++it) {
                ra[it] = *(const v8h*)&A [(rowBase + sr[it]) * (size_t)Kd + k0 + 32 + sc[it]];
                rb[it] = *(const v8h*)&BT[(size_t)sr[it] * Kd + k0 + 32 + sc[it]];
            }
            if (k0 + 64 < Kd)   // near-cache prefetch of tile k+2 of the streaming operand
                __builtin_prefetch(&A[(rowBase + sr[0]) * (size_t)Kd + k0 + 64 + sc[0]], 0, 3);
        }

        // ---- fragments per documented 16-bit A(16x32) / B(32x16) lane layouts ----
        union { v16h v; half_t h[16]; } afr[2], bfr[4];
#pragma unroll
        for (int mt = 0; mt < 2; ++mt) {
            const half_t* p = &As[buf][wm + mt * 16 + r16][0];
            *(v8h*)&afr[mt].h[0] = *(const v8h*)&p[koff];
            *(v8h*)&afr[mt].h[8] = *(const v8h*)&p[16 + koff];
        }
#pragma unroll
        for (int nt = 0; nt < 4; ++nt) {
            const half_t* p = &BsT[buf][wn + nt * 16 + r16][0];
            *(v8h*)&bfr[nt].h[0] = *(const v8h*)&p[koff];
            *(v8h*)&bfr[nt].h[8] = *(const v8h*)&p[16 + koff];
        }

#pragma unroll
        for (int mt = 0; mt < 2; ++mt)
#pragma unroll
            for (int nt = 0; nt < 4; ++nt)
                accf[mt][nt] = __builtin_amdgcn_wmma_f32_16x16x32_f16(
                    false, afr[mt].v, false, bfr[nt].v,
                    (short)0, accf[mt][nt], false, false);

        // ---- commit next tile into the other buffer ----
        if (hasNext) {
#pragma unroll
            for (int it = 0; it < 2; ++it) {
                *(v8h*)&As[buf ^ 1][sr[it]][sc[it]]  = ra[it];
                *(v8h*)&BsT[buf ^ 1][sr[it]][sc[it]] = rb[it];
            }
        }
        __syncthreads();
    }

    // ---- epilogue: C layout VGPR j -> M = j + (lane>=16 ? 8 : 0), N = lane&15 ----
    const int mAdd = (lane >> 4) * 8;
#pragma unroll
    for (int mt = 0; mt < 2; ++mt) {
        int dn[8];
        if (scatter) {
#pragma unroll
            for (int j = 0; j < 8; ++j)
                dn[j] = dstIdx[rowBase + wm + mt * 16 + j + mAdd];
        }
#pragma unroll
        for (int nt = 0; nt < 4; ++nt) {
            int colT = wn + nt * 16 + r16;
            if (colT >= Nreal) continue;
            float b = bias ? bias[colT] : 0.0f;
#pragma unroll
            for (int j = 0; j < 8; ++j) {
                float val = accf[mt][nt][j] + b;
                if (scatter) {
                    atomicAdd(&acc[(size_t)dn[j] * accLd + colT], val);
                } else {
                    if (relu) val = fmaxf(val, 0.0f);
                    Cout[(rowBase + wm + mt * 16 + j + mAdd) * H_DIM + colT] = (half_t)val;
                }
            }
        }
    }
}

// ---------------------------------------------------------------------------
// finalize scatter-mean: out[n,c] = acc[n,c]/max(deg,1) (+ residual)
// ---------------------------------------------------------------------------
__global__ void k_finalize(const float* __restrict__ acc, int accLd,
                           const float* __restrict__ deg,
                           const float* __restrict__ vres,
                           float* __restrict__ out, int outLd,
                           int Nc, int N_) {
    int idx = blockIdx.x * blockDim.x + threadIdx.x;
    if (idx >= N_ * Nc) return;
    int n = idx / Nc, c = idx % Nc;
    float d = fmaxf(deg[n], 1.0f);
    float val = acc[(size_t)n * accLd + c] / d;
    if (vres) val += vres[(size_t)n * Nc + c];
    out[(size_t)n * outLd + c] = val;
}

// ---------------------------------------------------------------------------
// BatchNorm: per-feature batch stats (biased var) over N rows of [N,128]
// ---------------------------------------------------------------------------
__global__ void k_bn_stats(const float* __restrict__ v, int N_,
                           float* __restrict__ mu, float* __restrict__ rinv) {
    int c = blockIdx.x;  // 0..127
    float s = 0.0f, s2 = 0.0f;
    for (int n = threadIdx.x; n < N_; n += blockDim.x) {
        float x = v[(size_t)n * H_DIM + c];
        s += x; s2 += x * x;
    }
    __shared__ float sh[256], sh2[256];
    sh[threadIdx.x] = s; sh2[threadIdx.x] = s2;
    __syncthreads();
    for (int o = 128; o > 0; o >>= 1) {
        if (threadIdx.x < o) {
            sh[threadIdx.x]  += sh[threadIdx.x + o];
            sh2[threadIdx.x] += sh2[threadIdx.x + o];
        }
        __syncthreads();
    }
    if (threadIdx.x == 0) {
        float m   = sh[0] / (float)N_;
        float var = sh2[0] / (float)N_ - m * m;
        mu[c]   = m;
        rinv[c] = rsqrtf(var + 1e-5f);
    }
}

__global__ void k_bn_apply(float* __restrict__ v, int N_,
                           const float* __restrict__ mu, const float* __restrict__ rinv,
                           const float* __restrict__ g, const float* __restrict__ b) {
    int idx = blockIdx.x * blockDim.x + threadIdx.x;
    if (idx >= N_ * H_DIM) return;
    int c = idx & (H_DIM - 1);
    v[idx] = (v[idx] - mu[c]) * rinv[c] * g[c] + b[c];
}

// ---------------------------------------------------------------------------
extern "C" void kernel_launch(void* const* d_in, const int* in_sizes, int n_in,
                              void* d_out, int out_size, void* d_ws, size_t ws_size,
                              hipStream_t stream) {
    const float* x   = (const float*)d_in[0];
    const float* ea  = (const float*)d_in[1];
    const int*   ei  = (const int*)d_in[2];
    // params pytree (dicts sorted by key): bn1{b,g} bn2{b,g} mp0..mp3{b0,b1,b2,w0,w1,w2}
    const float* bn1_b = (const float*)d_in[3];
    const float* bn1_g = (const float*)d_in[4];
    const float* bn2_b = (const float*)d_in[5];
    const float* bn2_g = (const float*)d_in[6];
    const float* b0[4], *b1[4], *b2[4], *w0[4], *w1[4], *w2[4];
    for (int l = 0; l < 4; ++l) {
        int base = 7 + l * 6;
        b0[l] = (const float*)d_in[base + 0];
        b1[l] = (const float*)d_in[base + 1];
        b2[l] = (const float*)d_in[base + 2];
        w0[l] = (const float*)d_in[base + 3];
        w1[l] = (const float*)d_in[base + 4];
        w2[l] = (const float*)d_in[base + 5];
    }

    const int N_ = in_sizes[0] / 16;       // 10000
    const int E_ = in_sizes[2] / 2;        // 320000 (multiple of 128)
    float* out = (float*)d_out;

    // ----- workspace carve-up -----
    char* ws = (char*)d_ws;
    size_t off = 0;
    auto carve = [&](size_t bytes) -> char* {
        char* p = ws + off;
        off = (off + bytes + 255) & ~(size_t)255;
        return p;
    };
    const int DinPad0 = 32, DinPad = 160;
    half_t* W0p[4]; half_t* W1p[4]; half_t* W2p[4];
    int k0pad[4] = {DinPad0, DinPad, DinPad, DinPad};
    int k0in [4] = {32, 144, 144, 144};
    int n2in [4] = {H_DIM, H_DIM, H_DIM, 3};
    for (int l = 0; l < 4; ++l) {
        W0p[l] = (half_t*)carve((size_t)k0pad[l] * H_DIM * sizeof(half_t));
        W1p[l] = (half_t*)carve((size_t)H_DIM * H_DIM * sizeof(half_t));
        W2p[l] = (half_t*)carve((size_t)H_DIM * H_DIM * sizeof(half_t));
    }
    half_t* bufA = (half_t*)carve((size_t)E_ * DinPad * sizeof(half_t));  // edge in / H2
    half_t* bufH = (half_t*)carve((size_t)E_ * H_DIM * sizeof(half_t));   // H1
    float*  acc  = (float*)carve((size_t)N_ * H_DIM * sizeof(float));
    float*  deg  = (float*)carve((size_t)N_ * sizeof(float));
    float*  v_a  = (float*)carve((size_t)N_ * H_DIM * sizeof(float));
    float*  v_b  = (float*)carve((size_t)N_ * H_DIM * sizeof(float));
    float*  mu   = (float*)carve(H_DIM * sizeof(float));
    float*  rinv = (float*)carve(H_DIM * sizeof(float));
    (void)ws_size; (void)n_in; (void)out_size;

    // ----- weight conversion (fp32 [K,N] -> transposed padded fp16 BT[128][KPad]) -----
    for (int l = 0; l < 4; ++l) {
        int t;
        t = k0pad[l] * H_DIM;
        k_convert_pad_T<<<(t + 255) / 256, 256, 0, stream>>>(w0[l], W0p[l], k0in[l], H_DIM, k0pad[l]);
        t = H_DIM * H_DIM;
        k_convert_pad_T<<<(t + 255) / 256, 256, 0, stream>>>(w1[l], W1p[l], H_DIM, H_DIM, H_DIM);
        k_convert_pad_T<<<(t + 255) / 256, 256, 0, stream>>>(w2[l], W2p[l], H_DIM, n2in[l], H_DIM);
    }

    // ----- degree -----
    hipMemsetAsync(deg, 0, (size_t)N_ * sizeof(float), stream);
    k_deg<<<(E_ + 255) / 256, 256, 0, stream>>>(ei, E_, deg);

    const int gemmGrid = E_ / 128;

    auto run_mp = [&](int l, const float* vin, int Dv, int dinPad,
                      const float* vres, float* vout, int outLd, int Nc, int accLd) {
        // edge input gather
        k_gather<<<E_, dinPad, 0, stream>>>(vin, Dv, ea, ei, E_, bufA, dinPad);
        // GEMM1: bufA[E,dinPad] x W0 -> relu -> bufH[E,128]
        k_gemm<<<gemmGrid, 256, 0, stream>>>(bufA, dinPad, W0p[l], H_DIM,
                                             b0[l], 1, 0, bufH, nullptr, 0, nullptr);
        // GEMM2: bufH x W1 -> relu -> bufA[E,128] (reuse)
        k_gemm<<<gemmGrid, 256, 0, stream>>>(bufH, H_DIM, W1p[l], H_DIM,
                                             b1[l], 1, 0, bufA, nullptr, 0, nullptr);
        // GEMM3: bufA x W2 -> +bias -> atomic scatter into acc[dst]
        hipMemsetAsync(acc, 0, (size_t)N_ * H_DIM * sizeof(float), stream);
        k_gemm<<<gemmGrid, 256, 0, stream>>>(bufA, H_DIM, W2p[l], Nc,
                                             b2[l], 0, 1, nullptr, acc, accLd, ei + E_);
        // scatter-mean (+ optional residual)
        int tot = N_ * Nc;
        k_finalize<<<(tot + 255) / 256, 256, 0, stream>>>(acc, accLd, deg, vres, vout, outLd, Nc, N_);
    };

    // layer 0: v_a = mp0(x)
    run_mp(0, x, 16, DinPad0, nullptr, v_a, H_DIM, H_DIM, H_DIM);
    // bn1 in place on v_a
    k_bn_stats<<<H_DIM, 256, 0, stream>>>(v_a, N_, mu, rinv);
    k_bn_apply<<<(N_ * H_DIM + 255) / 256, 256, 0, stream>>>(v_a, N_, mu, rinv, bn1_g, bn1_b);
    // layer 1: v_b = v_a + mp1(v_a)
    run_mp(1, v_a, H_DIM, DinPad, v_a, v_b, H_DIM, H_DIM, H_DIM);
    // bn2 in place on v_b
    k_bn_stats<<<H_DIM, 256, 0, stream>>>(v_b, N_, mu, rinv);
    k_bn_apply<<<(N_ * H_DIM + 255) / 256, 256, 0, stream>>>(v_b, N_, mu, rinv, bn2_g, bn2_b);
    // layer 2: v_a = v_b + mp2(v_b)
    run_mp(2, v_b, H_DIM, DinPad, v_b, v_a, H_DIM, H_DIM, H_DIM);
    // layer 3: out[N,3] = mp3(v_a)   (acc ld = 16, only first 3 cols written)
    run_mp(3, v_a, H_DIM, DinPad, nullptr, out, 3, 3, 16);
}
